// ED_LSTM_1D_11458972745749
// MI455X (gfx1250) — matrix operand order repaired and checked by reading, not compiled, not run on hardware
//
#include <hip/hip_runtime.h>
#include <hip/hip_bf16.h>

typedef __attribute__((ext_vector_type(16))) _Float16 v16h;
typedef __attribute__((ext_vector_type(8)))  float    v8f;

#define HDIM   64
#define G4     256
#define TSTEPS 1024
#define ENC    512
#define OUTW   559      // 512 encoder cols + 47 decoder cols
#define DECN   47
#define AIMW   600

// CDNA5 has a native v_tanh_f32 transcendental; use it when the toolchain
// exposes the builtin, else fall back to libm.
__device__ __forceinline__ float fast_tanh(float x) {
#if defined(__has_builtin)
#if __has_builtin(__builtin_amdgcn_tanhf)
  return __builtin_amdgcn_tanhf(x);
#else
  return tanhf(x);
#endif
#else
  return tanhf(x);
#endif
}
__device__ __forceinline__ float sigf(float x) {
  return 0.5f + 0.5f * fast_tanh(0.5f * x);
}

__device__ __forceinline__ v8f wmma16(v16h a, v16h b, v8f c) {
  // D = A(16x32 f16) * B(32x16 f16) + C(16x16 f32)
  return __builtin_amdgcn_wmma_f32_16x16x32_f16(false, a, false, b, (short)0, c, false, false);
}

// ---------------------------------------------------------------------------
// Pack weights into WMMA B-fragment order (f16) + fused bias into d_ws.
// B-frag slot = m*1024 + ntile*64 + kc*32 + lane, each slot = 16 halfs (32B).
// Lane l, half j of a 32x16 B tile holds W^T[K][N] with
//   K = kc*32 + (l<16 ? 0 : 16) + j,  N = l&15   (contiguous k order, same as the A frags).
// ntile = gate*4 + hidden-group; matrices m: 0..2 = w_hh[l], 3..4 = w_ih12[l].
// ---------------------------------------------------------------------------
__global__ void pack_kernel(const float* __restrict__ w_ih12,
                            const float* __restrict__ w_hh,
                            const float* __restrict__ b_ih,
                            const float* __restrict__ b_hh,
                            _Float16* __restrict__ bfrag,
                            float* __restrict__ bsum) {
  int idx = blockIdx.x * blockDim.x + threadIdx.x;
  if (idx < 5 * 1024) {
    int lane  = idx & 31;
    int kc    = (idx >> 5) & 1;
    int ntile = (idx >> 6) & 15;
    int m     = idx >> 10;                       // 0..4
    const float* W = (m < 3) ? (w_hh + m * G4 * HDIM)
                             : (w_ih12 + (m - 3) * G4 * HDIM);
    int gate = ntile >> 2, grp = ntile & 3;
    int col  = gate * 64 + grp * 16 + (lane & 15);   // gate row of W
    int kb   = kc * 32 + ((lane < 16) ? 0 : 16);
    _Float16 tmp[16];
#pragma unroll
    for (int j = 0; j < 16; ++j) tmp[j] = (_Float16)W[col * HDIM + kb + j];
    ((v16h*)bfrag)[idx] = *(const v16h*)tmp;
  }
  if (idx < 3 * G4) bsum[idx] = b_ih[idx] + b_hh[idx];
}

// ---------------------------------------------------------------------------
// Encoder: 32 WGs x 128 threads. WG b owns batch rows [16b, 16b+16).
// Wave w owns hidden group w (16 cols) x 4 gates -> 4 v8f accumulators.
// All 40 B fragments live in VGPRs for the whole recurrence; h state is
// double-buffered in LDS directly in WMMA A-fragment layout (f16); c state
// in VGPRs (fp32). 3 barriers per timestep.
// ---------------------------------------------------------------------------
__global__ __launch_bounds__(128) void enc_kernel(
    const float* __restrict__ aim_p, const float* __restrict__ w_ih0,
    const float* __restrict__ lin_w, const float* __restrict__ lin_b,
    const _Float16* __restrict__ bfragG, const float* __restrict__ bsum,
    float* __restrict__ out, float* __restrict__ wsH, float* __restrict__ wsC) {
  __shared__ __align__(32) _Float16 haFrag[2][3][2][32][16]; // [buf][layer][kc][lane][16]
  __shared__ float h3plain[16][HDIM];
  __shared__ float linw_s[HDIM];

  const int tid  = threadIdx.x;
  const int lane = tid & 31;
  const int grp  = tid >> 5;          // hidden group 0..3
  const int n    = lane & 15;
  const int rh   = lane >> 4;         // row-half of C-layout
  const int rowBase = blockIdx.x * 16;
  const bool lastBlk = (blockIdx.x == 31);

  for (int i = tid; i < 2 * 3 * 2 * 32 * 16; i += 128)
    ((_Float16*)haFrag)[i] = (_Float16)0.f;
  if (tid < HDIM) linw_s[tid] = lin_w[tid];
  __syncthreads();

  // per-lane constants (N = gate*64 + grp*16 + n is fixed per lane per gate)
  float bias0[4], bias1[4], bias2[4], w0v[4];
#pragma unroll
  for (int g = 0; g < 4; ++g) {
    int col = g * 64 + grp * 16 + n;
    bias0[g] = bsum[col];
    bias1[g] = bsum[G4 + col];
    bias2[g] = bsum[2 * G4 + col];
    w0v[g]   = w_ih0[col];
  }
  const float linb = lin_b[0];

  // A-fragment writeback mapping for hidden col hc (fixed per lane):
  const int hc  = grp * 16 + n;
  const int kcW = hc >> 5;
  const int Kin = hc & 31;
  const int sW  = Kin >> 4;
  const int jW  = Kin & 15;

  // Preload ALL B fragments into registers (loop-invariant weights).
  const v16h* BF = (const v16h*)bfragG;
  v16h B1h[4][2], B2h[4][2], B3h[4][2], B2x[4][2], B3x[4][2];
#pragma unroll
  for (int g = 0; g < 4; ++g) {
    const int nt = g * 4 + grp;
#pragma unroll
    for (int kc = 0; kc < 2; ++kc) {
      B1h[g][kc] = BF[0 * 1024 + nt * 64 + kc * 32 + lane];
      B2h[g][kc] = BF[1 * 1024 + nt * 64 + kc * 32 + lane];
      B3h[g][kc] = BF[2 * 1024 + nt * 64 + kc * 32 + lane];
      B2x[g][kc] = BF[3 * 1024 + nt * 64 + kc * 32 + lane];
      B3x[g][kc] = BF[4 * 1024 + nt * 64 + kc * 32 + lane];
    }
  }

  v8f cst0, cst1, cst2;
#pragma unroll
  for (int e = 0; e < 8; ++e) { cst0[e] = 0.f; cst1[e] = 0.f; cst2[e] = 0.f; }

  for (int t = 0; t < TSTEPS; ++t) {
    const int cur = t & 1, nxt = cur ^ 1;

    // per-lane input rows (contiguous, broadcast-friendly, cache-hot)
    const float* xrow = aim_p + t * AIMW + rowBase + rh * 8;
    float xv[8];
    *(float4*)&xv[0] = *(const float4*)xrow;
    *(float4*)&xv[4] = *(const float4*)(xrow + 4);
    if (t + 1 < TSTEPS) __builtin_prefetch(xrow + AIMW, 0, 0);

    // ---------- layer 1: gates = x*w_ih0 + h1 @ w_hh0^T + b ----------
    {
      v16h a0 = *(const v16h*)&haFrag[cur][0][0][lane][0];
      v16h a1 = *(const v16h*)&haFrag[cur][0][1][lane][0];
      v8f acc[4];
#pragma unroll
      for (int g = 0; g < 4; ++g) {
#pragma unroll
        for (int e = 0; e < 8; ++e) acc[g][e] = bias0[g] + xv[e] * w0v[g];
        acc[g] = wmma16(a0, B1h[g][0], acc[g]);
        acc[g] = wmma16(a1, B1h[g][1], acc[g]);
      }
#pragma unroll
      for (int e = 0; e < 8; ++e) {
        float iv = sigf(acc[0][e]), fv = sigf(acc[1][e]);
        float gv = fast_tanh(acc[2][e]), ov = sigf(acc[3][e]);
        float cn = fv * cst0[e] + iv * gv; cst0[e] = cn;
        float hv = ov * fast_tanh(cn);
        int row = rh * 8 + e;
        haFrag[nxt][0][kcW][row + 16 * sW][jW] = (_Float16)hv;
        if (t == TSTEPS - 1 && lastBlk && row == 15) { wsH[hc] = hv; wsC[hc] = cn; }
      }
      __syncthreads();   // h1(t) visible
    }

    // ---------- layer 2: gates = h1 @ w_ih1^T + h2 @ w_hh1^T + b ----------
    {
      v16h x0 = *(const v16h*)&haFrag[nxt][0][0][lane][0];
      v16h x1 = *(const v16h*)&haFrag[nxt][0][1][lane][0];
      v16h a0 = *(const v16h*)&haFrag[cur][1][0][lane][0];
      v16h a1 = *(const v16h*)&haFrag[cur][1][1][lane][0];
      v8f acc[4];
#pragma unroll
      for (int g = 0; g < 4; ++g) {
#pragma unroll
        for (int e = 0; e < 8; ++e) acc[g][e] = bias1[g];
        acc[g] = wmma16(x0, B2x[g][0], acc[g]);
        acc[g] = wmma16(x1, B2x[g][1], acc[g]);
        acc[g] = wmma16(a0, B2h[g][0], acc[g]);
        acc[g] = wmma16(a1, B2h[g][1], acc[g]);
      }
#pragma unroll
      for (int e = 0; e < 8; ++e) {
        float iv = sigf(acc[0][e]), fv = sigf(acc[1][e]);
        float gv = fast_tanh(acc[2][e]), ov = sigf(acc[3][e]);
        float cn = fv * cst1[e] + iv * gv; cst1[e] = cn;
        float hv = ov * fast_tanh(cn);
        int row = rh * 8 + e;
        haFrag[nxt][1][kcW][row + 16 * sW][jW] = (_Float16)hv;
        if (t == TSTEPS - 1 && lastBlk && row == 15) { wsH[HDIM + hc] = hv; wsC[HDIM + hc] = cn; }
      }
      __syncthreads();   // h2(t) visible
    }

    // ---------- layer 3: gates = h2 @ w_ih2^T + h3 @ w_hh2^T + b ----------
    {
      v16h x0 = *(const v16h*)&haFrag[nxt][1][0][lane][0];
      v16h x1 = *(const v16h*)&haFrag[nxt][1][1][lane][0];
      v16h a0 = *(const v16h*)&haFrag[cur][2][0][lane][0];
      v16h a1 = *(const v16h*)&haFrag[cur][2][1][lane][0];
      v8f acc[4];
#pragma unroll
      for (int g = 0; g < 4; ++g) {
#pragma unroll
        for (int e = 0; e < 8; ++e) acc[g][e] = bias2[g];
        acc[g] = wmma16(x0, B3x[g][0], acc[g]);
        acc[g] = wmma16(x1, B3x[g][1], acc[g]);
        acc[g] = wmma16(a0, B3h[g][0], acc[g]);
        acc[g] = wmma16(a1, B3h[g][1], acc[g]);
      }
#pragma unroll
      for (int e = 0; e < 8; ++e) {
        float iv = sigf(acc[0][e]), fv = sigf(acc[1][e]);
        float gv = fast_tanh(acc[2][e]), ov = sigf(acc[3][e]);
        float cn = fv * cst2[e] + iv * gv; cst2[e] = cn;
        float hv = ov * fast_tanh(cn);
        int row = rh * 8 + e;
        haFrag[nxt][2][kcW][row + 16 * sW][jW] = (_Float16)hv;
        h3plain[row][hc] = hv;
        if (t == TSTEPS - 1 && lastBlk && row == 15) { wsH[2 * HDIM + hc] = hv; wsC[2 * HDIM + hc] = cn; }
      }
      __syncthreads();   // h3(t) + h3plain visible
    }

    // ---------- projection: out[t, b] = h3 . lin_w + lin_b ----------
    if (tid < 16) {
      float o = linb;
#pragma unroll 8
      for (int k = 0; k < HDIM; ++k) o += h3plain[tid][k] * linw_s[k];
      out[t * OUTW + rowBase + tid] = o;
    }
  }
}

// ---------------------------------------------------------------------------
// Decoder: inherently serial (iteration d starts from iteration d-1's FINAL
// h,c), batch 1. One WG, one gate per thread, fp32 dots; the generated
// sequence cycles through LDS (obuf). c state lives in registers of the
// first 64 threads; projection is a wave0 shfl_xor reduction (wave32).
// ---------------------------------------------------------------------------
__global__ __launch_bounds__(256) void dec_kernel(
    const float* __restrict__ w_ih0, const float* __restrict__ w_ih12,
    const float* __restrict__ w_hh, const float* __restrict__ lin_w,
    const float* __restrict__ lin_b, const float* __restrict__ bsum,
    const float* __restrict__ wsH, const float* __restrict__ wsC,
    float* __restrict__ out) {
  __shared__ float obuf[TSTEPS];
  __shared__ float h[3][HDIM], gpre[G4];
  const int tid = threadIdx.x;

  for (int a = tid; a < TSTEPS; a += 256) obuf[a] = out[a * OUTW + (ENC - 1)];
  float c0r = 0.f, c1r = 0.f, c2r = 0.f;
  if (tid < HDIM) {
    h[0][tid] = wsH[tid];
    h[1][tid] = wsH[HDIM + tid];
    h[2][tid] = wsH[2 * HDIM + tid];
    c0r = wsC[tid];
    c1r = wsC[HDIM + tid];
    c2r = wsC[2 * HDIM + tid];
  }
  __syncthreads();

  const float linb = lin_b[0];
  const float bs0 = bsum[tid], bs1 = bsum[G4 + tid], bs2 = bsum[2 * G4 + tid];
  const float w0  = w_ih0[tid];
  const float* wr0 = w_hh + tid * HDIM;
  const float* wr1 = w_hh + G4 * HDIM + tid * HDIM;
  const float* wr2 = w_hh + 2 * G4 * HDIM + tid * HDIM;
  const float* wi1 = w_ih12 + tid * HDIM;
  const float* wi2 = w_ih12 + G4 * HDIM + tid * HDIM;
  const float lw_lo = (tid < 32) ? lin_w[tid] : 0.f;
  const float lw_hi = (tid < 32) ? lin_w[tid + 32] : 0.f;

  for (int d = 0; d < DECN; ++d) {
    for (int a = 0; a < TSTEPS; ++a) {
      float x = obuf[a];
      float g = bs0 + x * w0;
#pragma unroll 8
      for (int k = 0; k < HDIM; ++k) g += h[0][k] * wr0[k];
      gpre[tid] = g;
      __syncthreads();
      if (tid < HDIM) {
        float iv = sigf(gpre[tid]), fv = sigf(gpre[HDIM + tid]);
        float gv = fast_tanh(gpre[2 * HDIM + tid]), ov = sigf(gpre[3 * HDIM + tid]);
        c0r = fv * c0r + iv * gv;
        h[0][tid] = ov * fast_tanh(c0r);
      }
      __syncthreads();

      g = bs1;
#pragma unroll 8
      for (int k = 0; k < HDIM; ++k) g += h[0][k] * wi1[k] + h[1][k] * wr1[k];
      gpre[tid] = g;
      __syncthreads();
      if (tid < HDIM) {
        float iv = sigf(gpre[tid]), fv = sigf(gpre[HDIM + tid]);
        float gv = fast_tanh(gpre[2 * HDIM + tid]), ov = sigf(gpre[3 * HDIM + tid]);
        c1r = fv * c1r + iv * gv;
        h[1][tid] = ov * fast_tanh(c1r);
      }
      __syncthreads();

      g = bs2;
#pragma unroll 8
      for (int k = 0; k < HDIM; ++k) g += h[1][k] * wi2[k] + h[2][k] * wr2[k];
      gpre[tid] = g;
      __syncthreads();
      if (tid < HDIM) {
        float iv = sigf(gpre[tid]), fv = sigf(gpre[HDIM + tid]);
        float gv = fast_tanh(gpre[2 * HDIM + tid]), ov = sigf(gpre[3 * HDIM + tid]);
        c2r = fv * c2r + iv * gv;
        h[2][tid] = ov * fast_tanh(c2r);
      }
      __syncthreads();

      // projection by wave 0: wave32 shfl_xor reduction over 64 terms
      if (tid < 32) {
        float p = h[2][tid] * lw_lo + h[2][tid + 32] * lw_hi;
        p += __shfl_xor(p, 16, 32);
        p += __shfl_xor(p, 8, 32);
        p += __shfl_xor(p, 4, 32);
        p += __shfl_xor(p, 2, 32);
        p += __shfl_xor(p, 1, 32);
        if (tid == 0) {
          float o = p + linb;
          obuf[a] = o;                   // feeds iteration d+1
          out[a * OUTW + ENC + d] = o;
        }
      }
      __syncthreads();
    }
  }
}

extern "C" void kernel_launch(void* const* d_in, const int* in_sizes, int n_in,
                              void* d_out, int out_size, void* d_ws, size_t ws_size,
                              hipStream_t stream) {
  (void)in_sizes; (void)n_in; (void)out_size; (void)ws_size;
  const float* aim_p  = (const float*)d_in[2];
  const float* w_ih0  = (const float*)d_in[6];
  const float* w_ih12 = (const float*)d_in[7];
  const float* w_hh   = (const float*)d_in[8];
  const float* b_ih   = (const float*)d_in[9];
  const float* b_hh   = (const float*)d_in[10];
  const float* lin_w  = (const float*)d_in[11];
  const float* lin_b  = (const float*)d_in[12];
  float* out = (float*)d_out;

  char* ws = (char*)d_ws;
  _Float16* bfrag = (_Float16*)ws;                           // 5*1024 slots * 32B
  float* bsum = (float*)(ws + 5 * 1024 * 32);                // 768 f32
  float* wsH  = (float*)(ws + 5 * 1024 * 32 + 768 * 4);      // 192 f32
  float* wsC  = (float*)(ws + 5 * 1024 * 32 + 768 * 4 + 192 * 4);

  pack_kernel<<<24, 256, 0, stream>>>(w_ih12, w_hh, b_ih, b_hh, bfrag, bsum);
  enc_kernel<<<32, 128, 0, stream>>>(aim_p, w_ih0, lin_w, lin_b, bfrag, bsum,
                                     out, wsH, wsC);
  dec_kernel<<<1, 256, 0, stream>>>(w_ih0, w_ih12, w_hh, lin_w, lin_b, bsum,
                                    wsH, wsC, out);
}
